// Encoder_4904852652983
// MI455X (gfx1250) — compile-verified, matrix operand order checked by reading, not run on hardware
//
#include <hip/hip_runtime.h>
#include <math.h>

// Problem constants (from reference)
#define NN 50000
#define EE 800000
#define HH 4
#define DD 64
#define HD 256          // H*D
#define INFEAT 128
#define SLOPE 0.2f
#define BN_EPS 1e-3f

typedef __attribute__((ext_vector_type(16))) __bf16 v16bf;
typedef __attribute__((ext_vector_type(8)))  __bf16 v8bf;
typedef __attribute__((ext_vector_type(8)))  float  v8f;
typedef unsigned int v4u __attribute__((ext_vector_type(4)));
typedef int          v8i __attribute__((ext_vector_type(8)));
typedef int          v4i __attribute__((ext_vector_type(4)));

static __device__ __forceinline__ float lrelu(float x) { return x > 0.f ? x : SLOPE * x; }

// ---------------------------------------------------------------------------
// Elementwise helpers
// ---------------------------------------------------------------------------
__global__ void fill_f32_kernel(float* __restrict__ p, float v, int n) {
  int t = blockIdx.x * 256 + threadIdx.x;
  if (t < n) p[t] = v;
}

__global__ void log1p_bf16_kernel(const float* __restrict__ x, __bf16* __restrict__ hb, int n) {
  int t = blockIdx.x * 256 + threadIdx.x;
  if (t < n) hb[t] = (__bf16)__logf(x[t] + 1.0f);
}

__global__ void f32_to_bf16_kernel(const float* __restrict__ a, __bf16* __restrict__ b, int n) {
  int t = blockIdx.x * 256 + threadIdx.x;
  if (t < n) b[t] = (__bf16)a[t];
}

// ---------------------------------------------------------------------------
// Weight repack: f32 (K x Nout, row major) -> bf16 in CDNA5 WMMA B-operand
// layout.  Tile = 32(K) x 16(N).  Per ISA 7.12.2 (16-bit operands, wave32):
//   lanes 0-15  : n = nt*16+lane,    elems 0-7 -> K 0-7,  elems 8-15 -> K 16-23
//   lanes 16-31 : n = nt*16+lane-16, elems 0-7 -> K 8-15, elems 8-15 -> K 24-31
// Packed so each lane's 16 bf16 values are one contiguous 32B load.
// ---------------------------------------------------------------------------
__global__ void pack_w_bf16_kernel(const float* __restrict__ W, __bf16* __restrict__ Wpk,
                                   int K, int Nout) {
  int p = blockIdx.x * 256 + threadIdx.x;
  int total = K * Nout;
  if (p >= total) return;
  int ntiles = Nout >> 4;
  int tile   = p >> 9;       // 512 elems per 32x16 tile
  int within = p & 511;
  int l = within >> 4;       // lane 0..31
  int i = within & 15;       // element 0..15
  int kt = tile / ntiles;
  int nt = tile % ntiles;
  int n  = nt * 16 + (l & 15);
  int krel = (l < 16) ? ((i < 8) ? i : 8 + i)
                      : ((i < 8) ? 8 + i : 16 + i);
  int k = kt * 32 + krel;
  Wpk[p] = (__bf16)W[(size_t)k * Nout + n];
}

// ---------------------------------------------------------------------------
// WMMA GEMM, TDM + LDS-staged A, NT-way register blocking, fully unrolled K.
//   Block = 256 threads (8 waves), covers 32 rows (2 m-tiles) x NT*64 cols.
//   Wave w: m-tile (w>>2), n-tile group (w&3) of NT consecutive 16-col tiles.
//   A tile (32 x K bf16, <=16KB) is DMA'd into LDS once by the Tensor Data
//   Mover (wave 0 issues tensor_load_to_lds, waits TENSORcnt, then barrier);
//   tensor_dim1 = M so rows past M zero-fill (handles odd m-tile tail).
//   K is a compile-time constant (KSTEPS template param) so the k-loop is a
//   flat DAG: the scheduler hoists ds_load_b128 / global_load_b128 directly
//   into the v_wmma_f32_16x16x32_bf16 source registers (no rotation copies).
// ---------------------------------------------------------------------------
template <int NT, int KSTEPS>   // Nout = NT*64, K = KSTEPS*32
__global__ __launch_bounds__(256)
void wmma_gemm_lds_kernel(const __bf16* __restrict__ A, const __bf16* __restrict__ Wpk,
                          const float* __restrict__ bias, float* __restrict__ C,
                          int M) {
  constexpr int Nout   = NT * 64;
  constexpr int ntiles = Nout / 16;
  constexpr int K      = KSTEPS * 32;
  __shared__ __bf16 smem[32 * 256];   // 16KB max: 32 rows x K (K<=256)

  const int lane = threadIdx.x & 31;
  const int wave = threadIdx.x >> 5;
  const int mt_local = wave >> 2;     // 0..1
  const int ng       = wave & 3;      // 0..3

#if __has_builtin(__builtin_amdgcn_tensor_load_to_lds)
  if (wave == 0) {
    unsigned lds_base =
        (unsigned)(unsigned long long)(__attribute__((address_space(3))) __bf16*)smem;
    unsigned long long ga =
        (unsigned long long)A + (unsigned long long)blockIdx.x * 32ull * (unsigned)K * 2ull;
    v4u g0;
    g0.x = 1u;                                            // count=1 (valid user D#)
    g0.y = lds_base;                                      // lds_addr
    g0.z = (unsigned)(ga & 0xFFFFFFFFu);                  // global_addr[31:0]
    g0.w = (unsigned)((ga >> 32) & 0x1FFFFFFu) | (2u << 30); // ga[56:32] | type=2
    v8i g1;
    g1[0] = 1 << 16;                                      // data_size=1 (2 bytes)
    g1[1] = (int)((unsigned)(K & 0xFFFF) << 16);          // tensor_dim0[15:0]
    g1[2] = (int)(((unsigned)K >> 16) | ((unsigned)(M & 0xFFFF) << 16)); // dim0 hi | dim1 lo
    g1[3] = (int)((((unsigned)M >> 16) & 0xFFFF) | ((unsigned)(K & 0xFFFF) << 16)); // dim1 hi | tile_dim0=K
    g1[4] = 32;                                           // tile_dim1=32, tile_dim2=0
    g1[5] = K;                                            // tensor_dim0_stride[31:0]
    g1[6] = 0;
    g1[7] = 0;
    v4i gz4 = {0, 0, 0, 0};
    v8i gz8 = {0, 0, 0, 0, 0, 0, 0, 0};
    __builtin_amdgcn_tensor_load_to_lds(g0, g1, gz4, gz4, gz8, 0);
    __builtin_amdgcn_s_wait_tensorcnt(0);
  }
#else
  // Fallback: cooperative global->LDS copy (global_load_b128 + ds_store_b128)
  for (int idx = threadIdx.x; idx < 32 * (K / 8); idx += 256) {
    int r  = idx / (K / 8);
    int c8 = idx % (K / 8);
    int grow = blockIdx.x * 32 + r;
    v8bf v = {};
    if (grow < M) v = *(const v8bf*)(A + (size_t)grow * K + c8 * 8);
    *(v8bf*)(smem + r * K + c8 * 8) = v;
  }
#endif
  __syncthreads();

  const int half = lane >> 4;         // A-operand K-half select
  const int l16  = lane & 15;
  const __bf16* ap    = smem + (mt_local * 16 + l16) * K + half * 8;
  const __bf16* bbase = Wpk + (size_t)lane * 16;

  v8f acc[NT];
#pragma unroll
  for (int j = 0; j < NT; ++j) acc[j] = (v8f){};

#pragma unroll
  for (int kt = 0; kt < KSTEPS; ++kt) {
    v8bf a0 = *(const v8bf*)(ap + kt * 32);          // ds_load_b128
    v8bf a1 = *(const v8bf*)(ap + kt * 32 + 16);     // ds_load_b128
    v16bf av = __builtin_shufflevector(a0, a1, 0, 1, 2, 3, 4, 5, 6, 7,
                                               8, 9, 10, 11, 12, 13, 14, 15);
#pragma unroll
    for (int j = 0; j < NT; ++j) {
      v16bf bv = *(const v16bf*)(bbase + ((size_t)(kt * ntiles + ng * NT + j) << 9));
      acc[j] = __builtin_amdgcn_wmma_f32_16x16x32_bf16(
          false, av, false, bv, (short)0, acc[j], false, false);
    }
  }

  // C layout: VGPR r -> M = r (lanes 0-15) / 8+r (lanes 16-31); N = lane&15.
  const int mt = blockIdx.x * 2 + mt_local;
  if (mt < (M >> 4)) {                                // wave-uniform tail guard
#pragma unroll
    for (int j = 0; j < NT; ++j) {
      const int n  = (ng * NT + j) * 16 + l16;
      const float bv = bias ? bias[n] : 0.0f;
#pragma unroll
      for (int r = 0; r < 8; ++r) {
        const int m = mt * 16 + (half ? 8 + r : r);
        C[(size_t)m * Nout + n] = acc[j][r] + bv;
      }
    }
  }
}

// ---------------------------------------------------------------------------
// Edge pass 1: score_eh = attn_h . leaky_relu(fs[src] + fd[dst]); atomicMax m
// Float atomic max via sign-split int trick.
// ---------------------------------------------------------------------------
static __device__ __forceinline__ void atomic_max_f32(float* addr, float val) {
  if (val >= 0.f) atomicMax((int*)addr, __float_as_int(val));
  else            atomicMin((unsigned int*)addr, __float_as_uint(val));
}

__global__ __launch_bounds__(256)
void edge_score_kernel(const float* __restrict__ fs, const float* __restrict__ fd,
                       const float* __restrict__ attn,
                       const int* __restrict__ src, const int* __restrict__ dst,
                       float* __restrict__ score, float* __restrict__ m, int EH) {
  int t = blockIdx.x * 256 + threadIdx.x;
  if (t >= EH) return;
  int e = t >> 2, h = t & 3;
  int s = src[e], d = dst[e];
  const float4* ps = (const float4*)(fs + (size_t)s * HD + h * DD);
  const float4* pd = (const float4*)(fd + (size_t)d * HD + h * DD);
  const float4* pa = (const float4*)(attn + h * DD);
  float acc = 0.f;
#pragma unroll
  for (int q = 0; q < 16; ++q) {
    float4 a = ps[q], b = pd[q], w = pa[q];
    acc += lrelu(a.x + b.x) * w.x;
    acc += lrelu(a.y + b.y) * w.y;
    acc += lrelu(a.z + b.z) * w.z;
    acc += lrelu(a.w + b.w) * w.w;
  }
  score[t] = acc;
  atomic_max_f32(&m[(size_t)d * HH + h], acc);
}

__global__ void fix_m_kernel(float* __restrict__ m, int n) {
  int t = blockIdx.x * 256 + threadIdx.x;
  if (t < n) {
    float v = m[t];
    if (!__builtin_isfinite(v)) m[t] = 0.f;
  }
}

// Edge pass 2: ex = exp(score - m[dst]); den[dst] += ex  (score overwritten)
__global__ __launch_bounds__(256)
void edge_exp_kernel(float* __restrict__ score, const float* __restrict__ m,
                     const int* __restrict__ dst, float* __restrict__ den, int EH) {
  int t = blockIdx.x * 256 + threadIdx.x;
  if (t >= EH) return;
  int e = t >> 2, h = t & 3;
  int d = dst[e];
  float ex = __expf(score[t] - m[(size_t)d * HH + h]);
  score[t] = ex;
  atomicAdd(&den[(size_t)d * HH + h], ex);
}

// Edge pass 3: rout[dst] += fs[src] * alpha   (rout pre-seeded with residual)
__global__ __launch_bounds__(256)
void edge_scatter_kernel(const float* __restrict__ ex, const float* __restrict__ den,
                         const int* __restrict__ src, const int* __restrict__ dst,
                         const float* __restrict__ fs, float* __restrict__ rout, int EH) {
  int t = blockIdx.x * 256 + threadIdx.x;
  if (t >= EH) return;
  int e = t >> 2, h = t & 3;
  int s = src[e], d = dst[e];
  float alpha = ex[t] / fmaxf(den[(size_t)d * HH + h], 1e-9f);
  const float4* ps = (const float4*)(fs + (size_t)s * HD + h * DD);
  float* pr = rout + (size_t)d * HD + h * DD;
#pragma unroll
  for (int q = 0; q < 16; ++q) {
    float4 a = ps[q];
    atomicAdd(pr + 4 * q + 0, a.x * alpha);
    atomicAdd(pr + 4 * q + 1, a.y * alpha);
    atomicAdd(pr + 4 * q + 2, a.z * alpha);
    atomicAdd(pr + 4 * q + 3, a.w * alpha);
  }
}

// Head mean (N,H,D) -> (N,D) bf16
__global__ void head_mean_bf16_kernel(const float* __restrict__ r, __bf16* __restrict__ out, int n) {
  int t = blockIdx.x * 256 + threadIdx.x;
  if (t >= n) return;
  int node = t >> 6, d = t & 63;
  const float* p = r + (size_t)node * HD + d;
  out[t] = (__bf16)(0.25f * (p[0] + p[64] + p[128] + p[192]));
}

// BatchNorm stage 1: per-block partial sums over rows -> atomicAdd channel sums
__global__ __launch_bounds__(256)
void bn_partial_kernel(const float* __restrict__ h1, float* __restrict__ sum,
                       float* __restrict__ sumsq, int rows, int rowsPerBlock) {
  int c = threadIdx.x;
  int r0 = blockIdx.x * rowsPerBlock;
  int r1 = min(r0 + rowsPerBlock, rows);
  float s = 0.f, ss = 0.f;
  for (int r = r0; r < r1; ++r) {
    float v = h1[(size_t)r * HD + c];
    s += v; ss += v * v;
  }
  atomicAdd(&sum[c], s);
  atomicAdd(&sumsq[c], ss);
}

// BatchNorm stage 2: scale/shift per channel
__global__ void bn_final_kernel(const float* __restrict__ sum, const float* __restrict__ sumsq,
                                const float* __restrict__ gamma, const float* __restrict__ beta,
                                float* __restrict__ scale, float* __restrict__ shift) {
  int c = threadIdx.x;
  float inv = 1.0f / (float)NN;
  float mu  = sum[c] * inv;
  float var = sumsq[c] * inv - mu * mu;
  float sc  = gamma[c] * rsqrtf(var + BN_EPS);
  scale[c] = sc;
  shift[c] = beta[c] - mu * sc;
}

// BatchNorm stage 3: normalize + ReLU -> bf16
__global__ void bn_apply_bf16_kernel(const float* __restrict__ h1, const float* __restrict__ scale,
                                     const float* __restrict__ shift, __bf16* __restrict__ out, int n) {
  int t = blockIdx.x * 256 + threadIdx.x;
  if (t >= n) return;
  int c = t & 255;
  float v = h1[t] * scale[c] + shift[c];
  out[t] = (__bf16)fmaxf(v, 0.f);
}

// ---------------------------------------------------------------------------
// Host orchestration
// ---------------------------------------------------------------------------
static inline int cdiv(int a, int b) { return (a + b - 1) / b; }

extern "C" void kernel_launch(void* const* d_in, const int* in_sizes, int n_in,
                              void* d_out, int out_size, void* d_ws, size_t ws_size,
                              hipStream_t stream) {
  // Input order: x, src, dst, then params flattened in dict insertion order
  // (None leaves for layers 1/2 Wres/bres are dropped).
  const float* x   = (const float*)d_in[0];
  const int*   src = (const int*)d_in[1];
  const int*   dst = (const int*)d_in[2];
  const float* Wsrc[3] = {(const float*)d_in[3],  (const float*)d_in[10], (const float*)d_in[15]};
  const float* bsrc[3] = {(const float*)d_in[4],  (const float*)d_in[11], (const float*)d_in[16]};
  const float* Wdst[3] = {(const float*)d_in[5],  (const float*)d_in[12], (const float*)d_in[17]};
  const float* bdst[3] = {(const float*)d_in[6],  (const float*)d_in[13], (const float*)d_in[18]};
  const float* attn[3] = {(const float*)d_in[7],  (const float*)d_in[14], (const float*)d_in[19]};
  const float* Wres0 = (const float*)d_in[8];
  const float* bres0 = (const float*)d_in[9];
  const float* W1    = (const float*)d_in[20];
  const float* b1    = (const float*)d_in[21];
  const float* gamma = (const float*)d_in[22];
  const float* beta  = (const float*)d_in[23];
  const float* W2    = (const float*)d_in[24];
  const float* b2    = (const float*)d_in[25];

  // Workspace carve (256B aligned slices)
  char* wsp = (char*)d_ws;
  size_t off = 0;
  auto carve = [&](size_t bytes) -> void* {
    void* p = wsp + off;
    off += (bytes + 255) & ~(size_t)255;
    return p;
  };
  __bf16* hbf   = (__bf16*)carve((size_t)NN * HD * sizeof(__bf16));
  float*  fs    = (float*) carve((size_t)NN * HD * sizeof(float));
  float*  fd    = (float*) carve((size_t)NN * HD * sizeof(float));
  float*  rA    = (float*) carve((size_t)NN * HD * sizeof(float));
  float*  rB    = (float*) carve((size_t)NN * HD * sizeof(float));
  float*  score = (float*) carve((size_t)EE * HH * sizeof(float));
  float*  mbuf  = (float*) carve((size_t)NN * HH * sizeof(float));
  float*  den   = (float*) carve((size_t)NN * HH * sizeof(float));
  float*  bnbuf = (float*) carve(1024 * sizeof(float)); // sum|sumsq|scale|shift
  __bf16* wpk   = (__bf16*)carve((size_t)256 * 256 * sizeof(__bf16));
  // Reuse (free by the time they're needed):
  __bf16* hmeanb = (__bf16*)fs;   // (N,64) bf16
  float*  h1     = fd;            // (N,256) f32
  __bf16* hnbf   = hbf;           // (N,256) bf16

  const int EH = EE * HH;
  const int gemmBlocks = cdiv(NN / 16, 2);   // 2 m-tiles per block

  auto gemm = [&](const __bf16* A, const float* W, const float* bias, float* C,
                  int K, int Nout) {
    pack_w_bf16_kernel<<<cdiv(K * Nout, 256), 256, 0, stream>>>(W, wpk, K, Nout);
    if (Nout == 256) {
      if (K == 256)
        wmma_gemm_lds_kernel<4, 8><<<gemmBlocks, 256, 0, stream>>>(A, wpk, bias, C, NN);
      else if (K == 128)
        wmma_gemm_lds_kernel<4, 4><<<gemmBlocks, 256, 0, stream>>>(A, wpk, bias, C, NN);
      else  // K == 64
        wmma_gemm_lds_kernel<4, 2><<<gemmBlocks, 256, 0, stream>>>(A, wpk, bias, C, NN);
    } else {  // Nout == 64, K == 256
      wmma_gemm_lds_kernel<1, 8><<<gemmBlocks, 256, 0, stream>>>(A, wpk, bias, C, NN);
    }
  };

  // h0 = log(x+1), bf16
  log1p_bf16_kernel<<<cdiv(NN * INFEAT, 256), 256, 0, stream>>>(x, hbf, NN * INFEAT);

  float* rout = rA;
  const float* hcur = nullptr;
  for (int i = 0; i < 3; ++i) {
    const int K = (i == 0) ? INFEAT : HD;
    gemm(hbf, Wsrc[i], bsrc[i], fs, K, HD);
    gemm(hbf, Wdst[i], bdst[i], fd, K, HD);
    fill_f32_kernel<<<cdiv(NN * HH, 256), 256, 0, stream>>>(mbuf, -INFINITY, NN * HH);
    fill_f32_kernel<<<cdiv(NN * HH, 256), 256, 0, stream>>>(den, 0.0f, NN * HH);
    edge_score_kernel<<<cdiv(EH, 256), 256, 0, stream>>>(fs, fd, attn[i], src, dst, score, mbuf, EH);
    fix_m_kernel<<<cdiv(NN * HH, 256), 256, 0, stream>>>(mbuf, NN * HH);
    edge_exp_kernel<<<cdiv(EH, 256), 256, 0, stream>>>(score, mbuf, dst, den, EH);
    // Seed rout with residual (segment_sum starts at 0; adding res first is equivalent)
    if (i == 0) {
      gemm(hbf, Wres0, bres0, rout, INFEAT, HD);
    } else {
      (void)hipMemcpyAsync(rout, hcur, (size_t)NN * HD * sizeof(float),
                           hipMemcpyDeviceToDevice, stream);
    }
    edge_scatter_kernel<<<cdiv(EH, 256), 256, 0, stream>>>(score, den, src, dst, fs, rout, EH);
    if (i < 2) {
      f32_to_bf16_kernel<<<cdiv(NN * HD, 256), 256, 0, stream>>>(rout, hbf, NN * HD);
      hcur = rout;
      rout = (rout == rA) ? rB : rA;
    }
  }
  // rout now holds layer-2 output (N,H,D) incl. residual

  head_mean_bf16_kernel<<<cdiv(NN * DD, 256), 256, 0, stream>>>(rout, hmeanb, NN * DD);

  // FFN: h1 = hmean @ W1 + b1 (K=64 -> 2 WMMA k-steps)
  gemm(hmeanb, W1, b1, h1, DD, HD);

  // BatchNorm (batch statistics over N)
  fill_f32_kernel<<<cdiv(512, 256), 256, 0, stream>>>(bnbuf, 0.0f, 512);
  bn_partial_kernel<<<NN / 125, 256, 0, stream>>>(h1, bnbuf, bnbuf + 256, NN, 125);
  bn_final_kernel<<<1, 256, 0, stream>>>(bnbuf, bnbuf + 256, gamma, beta, bnbuf + 512, bnbuf + 768);
  bn_apply_bf16_kernel<<<cdiv(NN * HD, 256), 256, 0, stream>>>(h1, bnbuf + 512, bnbuf + 768,
                                                               hnbf, NN * HD);

  // Output GEMM: (N,256) @ (256,64) + b2 -> d_out (f32)
  gemm(hnbf, W2, b2, (float*)d_out, HD, DD);

  (void)in_sizes; (void)n_in; (void)out_size; (void)ws_size;
}
